// _ResChebGC_74406013436208
// MI455X (gfx1250) — matrix operand order repaired and checked by reading, not compiled, not run on hardware
//
#include <hip/hip_runtime.h>

// ---------------------------------------------------------------------------
// Fused Chebyshev graph conv for MI455X (gfx1250, wave32, WMMA).
// bf16 WMMA (16x16x32) for both the spatial mixing (T_k * x, N padded 21->32)
// and the channel GEMM (K=384 = [x | T1 x | T2 x] against W), fp32 accumulate.
// Memory-bound target: ~352 MB HBM traffic @ 23.3 TB/s (~15 us floor).
// ---------------------------------------------------------------------------

typedef __attribute__((ext_vector_type(4)))  unsigned int u32x4;
typedef __attribute__((ext_vector_type(2)))  unsigned int u32x2;
typedef __attribute__((ext_vector_type(4)))  float        f32x4;
typedef __attribute__((ext_vector_type(16))) __bf16       v16bf;
typedef __attribute__((ext_vector_type(8)))  float        v8f;

union Op16 {                 // one WMMA A/B operand (16 bf16 per lane)
  u32x4        q[2];
  unsigned int u[8];
  v16bf        v;
};

#define N_PTS 21
#define CIN   128
#define COUT  128
#define KORD  3
#define BT    8              // batches per block
#define ROWS  (BT * N_PTS)   // 168 dense (b,n) rows per block
#define RPAD  176            // padded to 11 M-tiles of 16
#define MT_N  11
#define PX    136            // bf16 row pitch of X/Y LDS (bank-skewed, 16B-aligned)
#define PXT   40             // m pitch of transposed X (16B-aligned rows)

// LDS layout (bytes)
#define OFF_W   0
#define SZ_W    (KORD * 4 * 8 * 32 * 8 * 4)   // 98304: pre-packed B-operands
#define OFF_X   (OFF_W + SZ_W)
#define SZ_XY   (RPAD * PX * 2)               // 47872
#define OFF_Y   (OFF_X + SZ_XY)
#define OFF_XT  (OFF_Y + SZ_XY)
#define SZ_XT   (BT * 128 * PXT * 2)          // 81920
#define OFF_T1  (OFF_XT + SZ_XT)
#define OFF_T2  (OFF_T1 + 448 * 4)
#define OFF_DI  (OFF_T2 + 448 * 4)
#define SMEM_BYTES (OFF_DI + 32 * 4)          // ~279.7 KB (< 320 KB WGP LDS)

// Native bf16 convert (RNE) — lowers to gfx1250 hardware cvt path.
__device__ __forceinline__ unsigned short f2bf(float f) {
  __bf16 h = (__bf16)f;
  return __builtin_bit_cast(unsigned short, h);
}
__device__ __forceinline__ unsigned int f2bf2(float lo, float hi) {
  return (unsigned int)f2bf(lo) | ((unsigned int)f2bf(hi) << 16);
}

__global__ __launch_bounds__(256) void cheb_fused_kernel(
    const float* __restrict__ x, const float* __restrict__ adj,
    const float* __restrict__ wgt, const float* __restrict__ bias,
    float* __restrict__ out) {
  extern __shared__ __align__(16) char smem[];
  unsigned int*   wlds = (unsigned int*)(smem + OFF_W);
  unsigned short* xl   = (unsigned short*)(smem + OFF_X);
  unsigned short* yl   = (unsigned short*)(smem + OFF_Y);
  unsigned short* xt   = (unsigned short*)(smem + OFF_XT);
  float* T1 = (float*)(smem + OFF_T1);
  float* T2 = (float*)(smem + OFF_T2);
  float* di = (float*)(smem + OFF_DI);

  const int t    = threadIdx.x;
  const int lane = t & 31;
  const int wv   = t >> 5;        // wave id 0..7: owns batch wv (mix) and nt=wv (GEMM)
  const int blk  = blockIdx.x;

  const float* xg = x + (size_t)blk * (ROWS * CIN);

  // ---- kick HBM early: prefetch this block's x tile (672 x 128B lines) ----
  for (int i = t; i < (ROWS * CIN * 4) / 128; i += 256)
    __builtin_prefetch(xg + i * 32, 0, 0);   // -> global_prefetch_b8

  // ---- zero X / Y / XT (pad rows & pad m-columns must be 0 for WMMA) ----
  {
    u32x4 z = {};
    u32x4* p = (u32x4*)(smem + OFF_X);
    const int n16 = (2 * SZ_XY + SZ_XT) / 16;
    for (int i = t; i < n16; i += 256) p[i] = z;
  }
  // ---- D^{-1/2} from adj rowsums ----
  if (t < N_PTS) {
    float s = 0.f;
    for (int j = 0; j < N_PTS; ++j) s += adj[t * N_PTS + j];
    di[t] = (s > 0.f) ? rsqrtf(s) : 0.f;
  }
  __syncthreads();
  // ---- T1 = L = I - D^{-1/2} A D^{-1/2} ----
  if (t < N_PTS * N_PTS) {
    int n = t / N_PTS, m = t - n * N_PTS;
    T1[t] = ((n == m) ? 1.f : 0.f) - di[n] * adj[t] * di[m];
  }
  // ---- pack W (fp32->bf16 pairs) into LDS in exact B-operand layout ----
  for (int p = t; p < KORD * 64 * 128; p += 256) {
    int d  = p & 127;
    int c2 = (p >> 7) & 63;
    int k  = p >> 13;
    int c  = c2 * 2;
    float g0 = wgt[(k * CIN + c) * COUT + d];        // coalesced in d
    float g1 = wgt[(k * CIN + c + 1) * COUT + d];
    int ct = c >> 5;                 // 32-wide K tile within the 128-chunk
    int Li = ((c >> 4) & 1) * 16 + (d & 15);
    int v  = (c & 15) >> 1;
    int nt = d >> 4;
    wlds[(((k * 4 + ct) * 8 + nt) * 32 + Li) * 8 + v] = f2bf2(g0, g1);
  }
  // ---- stage x: row-major (A operand) + m-transposed (mix B operand) ----
  for (int i = 0; i < 21; ++i) {
    int idx4 = t + i * 256;                 // 5376 float4 per block
    int flat = idx4 * 4;
    int row  = flat >> 7;
    int col  = flat & 127;
    f32x4 vv = *(const f32x4*)(xg + flat);
    unsigned short h0 = f2bf(vv.x), h1 = f2bf(vv.y);
    unsigned short h2 = f2bf(vv.z), h3 = f2bf(vv.w);
    u32x2 pr;
    pr.x = (unsigned int)h0 | ((unsigned int)h1 << 16);
    pr.y = (unsigned int)h2 | ((unsigned int)h3 << 16);
    *(u32x2*)(xl + row * PX + col) = pr;
    int bb = row / N_PTS;
    int m  = row - bb * N_PTS;
    xt[(bb * 128 + col + 0) * PXT + m] = h0;
    xt[(bb * 128 + col + 1) * PXT + m] = h1;
    xt[(bb * 128 + col + 2) * PXT + m] = h2;
    xt[(bb * 128 + col + 3) * PXT + m] = h3;
  }
  __syncthreads();
  // ---- T2 = 2 L^2 - I ----
  if (t < N_PTS * N_PTS) {
    int n = t / N_PTS, m = t - n * N_PTS;
    float s = 0.f;
    for (int j = 0; j < N_PTS; ++j) s += T1[n * N_PTS + j] * T1[j * N_PTS + m];
    T2[t] = 2.f * s - ((n == m) ? 1.f : 0.f);
  }
  __syncthreads();

  // ---- build T1/T2 16x32 A-operands in registers (ISA A layout:
  //      lane L holds row L&15, VGPR v holds K pair (v>>2)*16+(L>>4)*8+(v&3)*2)
  Op16 tp[2][2];
  for (int kk = 0; kk < 2; ++kk)
    for (int mt = 0; mt < 2; ++mt) {
      const float* Tm = kk ? T2 : T1;
      int row = mt * 16 + (lane & 15);
      #pragma unroll
      for (int v = 0; v < 8; ++v) {
        int m0 = (v >> 2) * 16 + (lane >> 4) * 8 + (v & 3) * 2;
        float f0 = (row < N_PTS && m0     < N_PTS) ? Tm[row * N_PTS + m0]     : 0.f;
        float f1 = (row < N_PTS && m0 + 1 < N_PTS) ? Tm[row * N_PTS + m0 + 1] : 0.f;
        tp[kk][mt].u[v] = f2bf2(f0, f1);
      }
    }
  float bv = bias[wv * 16 + (lane & 15)];

  // ---- spatial mix via WMMA: y_k[b] = T_k * x[b] (padded 32x128) ----
  auto mix = [&](int kk) {
    const int b = wv;
    for (int nt = 0; nt < 8; ++nt) {
      int o = (b * 128 + nt * 16 + (lane & 15)) * PXT + (lane >> 4) * 8;
      Op16 B;
      B.q[0] = *(const u32x4*)(xt + o);
      B.q[1] = *(const u32x4*)(xt + o + 16);
      #pragma unroll
      for (int mt = 0; mt < 2; ++mt) {
        v8f d = {};
        d = __builtin_amdgcn_wmma_f32_16x16x32_bf16(
            false, tp[kk][mt].v, false, B.v, (short)0, d, false, false);
        int rbase = mt * 16 + ((lane >> 4) << 3);   // D layout: M = r + 8*(lane>=16)
        #pragma unroll
        for (int r = 0; r < 8; ++r) {
          int rib = rbase + r;
          if (rib < N_PTS)
            yl[(b * N_PTS + rib) * PX + nt * 16 + (lane & 15)] = f2bf(d[r]);
        }
      }
    }
  };

  v8f acc[MT_N];
  #pragma unroll
  for (int i = 0; i < MT_N; ++i) acc[i] = {};

  // ---- channel GEMM chunk: acc += A(src rows, 128-wide K chunk) x W_k ----
  auto gemm = [&](const unsigned short* src, int k) {
    for (int ct = 0; ct < 4; ++ct) {
      Op16 B;
      const u32x4* wb = (const u32x4*)wlds + (((k * 4 + ct) * 8 + wv) * 32 + lane) * 2;
      B.q[0] = wb[0];
      B.q[1] = wb[1];
      #pragma unroll
      for (int mt = 0; mt < MT_N; ++mt) {
        int off = (mt * 16 + (lane & 15)) * PX + ct * 32 + ((lane >> 4) << 4);
        Op16 A;
        A.q[0] = *(const u32x4*)(src + off);
        A.q[1] = *(const u32x4*)(src + off + 8);
        acc[mt] = __builtin_amdgcn_wmma_f32_16x16x32_bf16(
            false, A.v, false, B.v, (short)0, acc[mt], false, false);
      }
    }
  };

  mix(0);                    // y1 = T1 x -> yl
  __syncthreads();
  gemm(xl, 0);               // k=0: identity chunk (x itself)
  gemm(yl, 1);               // k=1
  __syncthreads();           // all waves done reading yl
  mix(1);                    // y2 = T2 x -> yl (reads only xt)
  __syncthreads();
  gemm(yl, 2);               // k=2

  // ---- bias + store fp32 output ----
  float* og = out + (size_t)blk * (ROWS * COUT);
  #pragma unroll
  for (int mt = 0; mt < MT_N; ++mt) {
    int rbase = mt * 16 + ((lane >> 4) << 3);
    #pragma unroll
    for (int r = 0; r < 8; ++r) {
      int rg = rbase + r;
      if (rg < ROWS)
        og[rg * COUT + wv * 16 + (lane & 15)] = acc[mt][r] + bv;
    }
  }
}

extern "C" void kernel_launch(void* const* d_in, const int* in_sizes, int n_in,
                              void* d_out, int out_size, void* d_ws, size_t ws_size,
                              hipStream_t stream) {
  const float* x    = (const float*)d_in[0];
  const float* adj  = (const float*)d_in[1];
  const float* wgt  = (const float*)d_in[2];
  const float* bias = (const float*)d_in[3];
  float* out = (float*)d_out;

  const int batch  = in_sizes[0] / (N_PTS * CIN);   // 16384
  const int blocks = batch / BT;                    // 2048

  (void)hipFuncSetAttribute(reinterpret_cast<const void*>(cheb_fused_kernel),
                            hipFuncAttributeMaxDynamicSharedMemorySize,
                            SMEM_BYTES);
  cheb_fused_kernel<<<dim3(blocks), dim3(256), SMEM_BYTES, stream>>>(
      x, adj, wgt, bias, out);
}
